// QuadraticLorentzAttention_45157286150964
// MI455X (gfx1250) — matrix operand order repaired and checked by reading, not compiled, not run on hardware
//
#include <hip/hip_runtime.h>

typedef __attribute__((ext_vector_type(16))) _Float16 v16h;
typedef __attribute__((ext_vector_type(8)))  _Float16 v8h;
typedef __attribute__((ext_vector_type(8)))  float    v8f;
typedef __attribute__((ext_vector_type(4)))  unsigned int u32x4;
typedef __attribute__((ext_vector_type(8)))  int          i32x8;
typedef __attribute__((ext_vector_type(4)))  int          i32x4;

// Problem constants (match reference setup_inputs)
#define NSEQ  2048
#define DL    65            // Lorentz coords (1 time + 64 space)
#define BH    32            // B*H
#define KSTRH 104           // K row stride (halves): 96 data (padded from 65) + 8, 16B rows
#define VSTRH 40            // V^T LDS row stride (halves): 32 data + 8 pad (TDM pad fields)
#define LOG2E 1.44269504088896340736f

#define K16_ELEMS ((size_t)BH * NSEQ * KSTRH)
#define V16_ELEMS ((size_t)BH * 64 * NSEQ)
#define WS_NEED   ((K16_ELEMS + V16_ELEMS) * sizeof(_Float16))

union AFrag { v16h v; v8h h[2]; };

// ============================================================================
// Prep: one-time f32 -> f16 conversion into TDM-friendly layouts.
//   K16 : [bh][n][104]  (k<65 data, else 0; row == LDS row the B-frags read)
//   V16T: [bh][d][2048] (space dims transposed so V B-frags are contiguous)
// ============================================================================
__global__ __launch_bounds__(256)
void lorentz_prep_f16(const float* __restrict__ Kg, const float* __restrict__ Vg,
                      _Float16* __restrict__ K16, _Float16* __restrict__ V16T)
{
    __shared__ _Float16 Vt[64][34];
    const int bh = blockIdx.y;
    const int r0 = blockIdx.x * 32;              // 32 sequence rows per block
    const float* Kh = Kg + (size_t)bh * NSEQ * DL;
    const float* Vh = Vg + (size_t)bh * NSEQ * DL;
    _Float16* Kd = K16  + (size_t)bh * NSEQ * KSTRH;
    _Float16* Vd = V16T + (size_t)bh * 64 * NSEQ;

    for (int idx = threadIdx.x; idx < 32 * KSTRH; idx += 256) {
        int r = idx / KSTRH, k = idx - r * KSTRH;
        float v = (k < DL) ? Kh[(size_t)(r0 + r) * DL + k] : 0.0f;
        Kd[(size_t)(r0 + r) * KSTRH + k] = (_Float16)v;
    }
    for (int idx = threadIdx.x; idx < 32 * 64; idx += 256) {
        int r = idx >> 6, d = idx & 63;
        Vt[d][r] = (_Float16)Vh[(size_t)(r0 + r) * DL + 1 + d];
    }
    __syncthreads();
    for (int idx = threadIdx.x; idx < 64 * 32; idx += 256) {   // coalesced transposed store
        int d = idx >> 5, n = idx & 31;
        Vd[(size_t)d * NSEQ + r0 + n] = Vt[d][n];
    }
}

// ============================================================================
// TDM 2-D tile load: build D# group0 (count/lds_addr/global_addr/type=2) and
// issue TENSOR_LOAD_TO_LDS with a precomputed group1. Groups 2/3 unused (2-D).
// ============================================================================
__device__ __forceinline__ void tdm_load_2d(unsigned lds_addr,
                                            unsigned long long gaddr,
                                            i32x8 g1)
{
    u32x4 g0;
    g0.x = 1u;                                            // count=1, user descriptor
    g0.y = lds_addr;                                      // LDS byte address
    g0.z = (unsigned)gaddr;                               // global_addr[31:0]
    g0.w = ((unsigned)(gaddr >> 32) & 0x01FFFFFFu) | 0x80000000u;  // [56:32] | type=2
    i32x4 z4 = {0, 0, 0, 0};
    i32x8 z8 = {0, 0, 0, 0, 0, 0, 0, 0};
    __builtin_amdgcn_tensor_load_to_lds(g0, g1, z4, z4, z8, 0);
}

// ============================================================================
// Flash attention, TDM double-buffered K/V tiles (f16 in, f32 WMMA accum)
// ============================================================================
__global__ __launch_bounds__(256)
void lorentz_flash_attn_tdm(const float*    __restrict__ Qg,
                            const _Float16* __restrict__ K16,
                            const _Float16* __restrict__ V16T,
                            const int*      __restrict__ causalp,
                            float*          __restrict__ Og)
{
    __shared__ __align__(16) _Float16 KshD[2][32 * KSTRH];
    __shared__ __align__(16) _Float16 VshD[2][64 * VSTRH];
    __shared__ __align__(16) _Float16 Psh[8][16 * 32];

    const int tid    = threadIdx.x;
    const int wid    = tid >> 5;
    const int lane   = tid & 31;
    const int laneN  = lane & 15;
    const int hiHalf = lane >> 4;
    const int m0     = blockIdx.x * 128;
    const int bh     = blockIdx.y;
    const int qBase  = m0 + wid * 16;
    const bool causal = (*causalp) != 0;

    const float* Qh = Qg + (size_t)bh * NSEQ * DL;
    float*       Oh = Og + (size_t)bh * NSEQ * DL;

    // ---- Q A-fragments: time negated, pre-scaled by log2(e) ----
    AFrag Qa[3];
    {
        const float* qrow = Qh + (size_t)(qBase + laneN) * DL;
        #pragma unroll
        for (int c = 0; c < 3; ++c) {
            #pragma unroll
            for (int i = 0; i < 16; ++i) {
                int k = c * 32 + i + ((i >= 8) ? 8 : 0) + hiHalf * 8;
                float v = (k < DL) ? qrow[k] : 0.0f;
                if (k == 0) v = -v;
                Qa[c].v[i] = (_Float16)(v * LOG2E);
            }
        }
    }
    AFrag OnesB;
    #pragma unroll
    for (int i = 0; i < 16; ++i) OnesB.v[i] = (_Float16)1.0f;

    v8f Oacc[4];
    #pragma unroll
    for (int dt = 0; dt < 4; ++dt) Oacc[dt] = (v8f){};
    v8f Lacc = (v8f){};

    // ---- TDM descriptors (group1 constant; group0 per tile) ----
    // K: 2-D tensor rows of 104 halves; tile 104x32, contiguous LDS rows.
    const i32x8 Kg1 = { 0x00010000,          // data_size=2B
                        (int)(KSTRH << 16),  // tensor_dim0 lo16
                        (int)(NSEQ  << 16),  // tensor_dim0 hi16 | tensor_dim1 lo16
                        (int)(KSTRH << 16),  // tensor_dim1 hi16 | tile_dim0
                        32,                  // tile_dim1 | tile_dim2=0
                        KSTRH, 0, 0 };       // tensor_dim0_stride
    // V^T: rows of 2048 halves; tile 32x64; LDS pad 4 dwords every 16 dwords -> stride 40.
    const i32x8 Vg1 = { 0x06D10000,          // 2B | pad_en | interval=16dw | amount=4dw
                        (int)(NSEQ << 16),
                        (int)(64   << 16),
                        (int)(32   << 16),
                        64,
                        NSEQ, 0, 0 };

    const unsigned ldsK = (unsigned)(size_t)(void*)&KshD[0][0];
    const unsigned ldsV = (unsigned)(size_t)(void*)&VshD[0][0];
    const unsigned long long kBase = (unsigned long long)(size_t)K16
                                   + (unsigned long long)bh * NSEQ * KSTRH * 2ull;
    const unsigned long long vBase = (unsigned long long)(size_t)V16T
                                   + (unsigned long long)bh * 64 * NSEQ * 2ull;

    const int nIter = causal ? ((m0 + 128) >> 5) : (NSEQ >> 5);

    auto issueTile = [&](int itx) {
        const int jj = itx * 32;
        const unsigned b = (unsigned)(itx & 1);
        tdm_load_2d(ldsK + b * (32 * KSTRH * 2u),
                    kBase + (unsigned long long)jj * (KSTRH * 2ull), Kg1);
        tdm_load_2d(ldsV + b * (64 * VSTRH * 2u),
                    vBase + (unsigned long long)jj * 2ull, Vg1);
    };

    if (wid == 0) issueTile(0);

    for (int it = 0; it < nIter; ++it) {
        const int j0 = it * 32;
        if (wid == 0) {
            if (it + 1 < nIter) {                 // prefetch next tile, wait for current
                issueTile(it + 1);
                __builtin_amdgcn_s_wait_tensorcnt(2);
            } else {
                __builtin_amdgcn_s_wait_tensorcnt(0);
            }
        }
        __syncthreads();                          // publish tile to all waves

        const _Float16* Ksh = KshD[it & 1];
        const _Float16* Vsh = VshD[it & 1];

        if (!(causal && j0 > qBase + 15)) {       // wave-uniform fully-masked skip
            const bool needMask = causal && (j0 + 31 > qBase);

            v8f S0 = (v8f){};
            v8f S1 = (v8f){};
            #pragma unroll
            for (int c = 0; c < 3; ++c) {
                AFrag B0, B1;
                const int ko = c * 32 + 8 * hiHalf;
                B0.h[0] = *(const v8h*)&Ksh[laneN * KSTRH + ko];
                B0.h[1] = *(const v8h*)&Ksh[laneN * KSTRH + ko + 16];
                B1.h[0] = *(const v8h*)&Ksh[(16 + laneN) * KSTRH + ko];
                B1.h[1] = *(const v8h*)&Ksh[(16 + laneN) * KSTRH + ko + 16];
                S0 = __builtin_amdgcn_wmma_f32_16x16x32_f16(false, Qa[c].v, false, B0.v,
                                                            (short)0, S0, false, false);
                S1 = __builtin_amdgcn_wmma_f32_16x16x32_f16(false, Qa[c].v, false, B1.v,
                                                            (short)0, S1, false, false);
            }

            // P = exp2(S): hyperboloid scores <= -1, so no running max needed
            #pragma unroll
            for (int r = 0; r < 8; ++r) {
                float s0 = S0[r], s1 = S1[r];
                if (needMask) {
                    const int qi = qBase + r + hiHalf * 8;
                    if (j0 + laneN      > qi) s0 = -3.0e38f;
                    if (j0 + 16 + laneN > qi) s1 = -3.0e38f;
                }
                const float p0 = __builtin_amdgcn_exp2f(s0);
                const float p1 = __builtin_amdgcn_exp2f(s1);
                const int m = r + hiHalf * 8;
                Psh[wid][m * 32 + laneN]      = (_Float16)p0;
                Psh[wid][m * 32 + 16 + laneN] = (_Float16)p1;
            }

            AFrag Pa;
            Pa.h[0] = *(const v8h*)&Psh[wid][laneN * 32 + 8 * hiHalf];
            Pa.h[1] = *(const v8h*)&Psh[wid][laneN * 32 + 16 + 8 * hiHalf];

            Lacc = __builtin_amdgcn_wmma_f32_16x16x32_f16(false, Pa.v, false, OnesB.v,
                                                          (short)0, Lacc, false, false);
            #pragma unroll
            for (int dt = 0; dt < 4; ++dt) {
                AFrag Vb;
                const int vr = (dt * 16 + laneN) * VSTRH + 8 * hiHalf;
                Vb.h[0] = *(const v8h*)&Vsh[vr];
                Vb.h[1] = *(const v8h*)&Vsh[vr + 16];
                Oacc[dt] = __builtin_amdgcn_wmma_f32_16x16x32_f16(false, Pa.v, false, Vb.v,
                                                                  (short)0, Oacc[dt],
                                                                  false, false);
            }
        }
        __syncthreads();                          // tile consumed: safe to overwrite
    }

    // ---- epilogue ----
    #pragma unroll
    for (int r = 0; r < 8; ++r) {
        const int qi  = qBase + r + hiHalf * 8;
        const float inv = 1.0f / Lacc[r];
        float y[4];
        float ss = 0.0f;
        #pragma unroll
        for (int dt = 0; dt < 4; ++dt) {
            y[dt] = Oacc[dt][r] * inv;
            ss += y[dt] * y[dt];
        }
        #pragma unroll
        for (int sft = 1; sft < 16; sft <<= 1)
            ss += __shfl_xor(ss, sft, 32);
        const float t = sqrtf(ss + 1.0f);
        float* orow = Oh + (size_t)qi * DL;
        #pragma unroll
        for (int dt = 0; dt < 4; ++dt)
            orow[1 + dt * 16 + laneN] = y[dt];
        if (laneN == 0) orow[0] = t;
    }
}

// ============================================================================
// Fallback (self-staging) path, used when d_ws is too small for the f16 copies
// ============================================================================
__global__ __launch_bounds__(256)
void lorentz_flash_attn_fallback(const float* __restrict__ Qg,
                                 const float* __restrict__ Kg,
                                 const float* __restrict__ Vg,
                                 const int*   __restrict__ causalp,
                                 float*       __restrict__ Og)
{
    __shared__ __align__(16) _Float16 KshH[32 * KSTRH];
    __shared__ __align__(16) _Float16 VshT[64 * VSTRH];
    __shared__ __align__(16) _Float16 Psh[8][16 * 32];

    const int tid    = threadIdx.x;
    const int wid    = tid >> 5;
    const int lane   = tid & 31;
    const int laneN  = lane & 15;
    const int hiHalf = lane >> 4;
    const int m0     = blockIdx.x * 128;
    const int bh     = blockIdx.y;
    const int qBase  = m0 + wid * 16;
    const bool causal = (*causalp) != 0;

    const size_t headOff = (size_t)bh * NSEQ * DL;
    const float* Qh = Qg + headOff;
    const float* Kh = Kg + headOff;
    const float* Vh = Vg + headOff;
    float*       Oh = Og + headOff;

    AFrag Qa[3];
    {
        const float* qrow = Qh + (size_t)(qBase + laneN) * DL;
        #pragma unroll
        for (int c = 0; c < 3; ++c) {
            #pragma unroll
            for (int i = 0; i < 16; ++i) {
                int k = c * 32 + i + ((i >= 8) ? 8 : 0) + hiHalf * 8;
                float v = (k < DL) ? qrow[k] : 0.0f;
                if (k == 0) v = -v;
                Qa[c].v[i] = (_Float16)(v * LOG2E);
            }
        }
    }
    AFrag OnesB;
    #pragma unroll
    for (int i = 0; i < 16; ++i) OnesB.v[i] = (_Float16)1.0f;

    v8f Oacc[4];
    #pragma unroll
    for (int dt = 0; dt < 4; ++dt) Oacc[dt] = (v8f){};
    v8f Lacc = (v8f){};

    const int nIter = causal ? ((m0 + 128) >> 5) : (NSEQ >> 5);

    for (int it = 0; it < nIter; ++it) {
        const int j0 = it * 32;
        __syncthreads();
        for (int idx = tid; idx < 32 * 96; idx += 256) {
            int r = idx / 96, k = idx - r * 96;
            float v = (k < DL) ? Kh[(size_t)(j0 + r) * DL + k] : 0.0f;
            KshH[r * KSTRH + k] = (_Float16)v;
        }
        for (int idx = tid; idx < 32 * 64; idx += 256) {
            int r = idx >> 6, d = idx & 63;
            VshT[d * VSTRH + r] = (_Float16)Vh[(size_t)(j0 + r) * DL + 1 + d];
        }
        if (wid == 0 && j0 + 63 < NSEQ)
            __builtin_prefetch(&Kh[(size_t)(j0 + 32 + lane) * DL], 0, 3);
        __syncthreads();

        if (causal && j0 > qBase + 15) continue;
        const bool needMask = causal && (j0 + 31 > qBase);

        v8f S0 = (v8f){};
        v8f S1 = (v8f){};
        #pragma unroll
        for (int c = 0; c < 3; ++c) {
            AFrag B0, B1;
            const int ko = c * 32 + 8 * hiHalf;
            B0.h[0] = *(const v8h*)&KshH[laneN * KSTRH + ko];
            B0.h[1] = *(const v8h*)&KshH[laneN * KSTRH + ko + 16];
            B1.h[0] = *(const v8h*)&KshH[(16 + laneN) * KSTRH + ko];
            B1.h[1] = *(const v8h*)&KshH[(16 + laneN) * KSTRH + ko + 16];
            S0 = __builtin_amdgcn_wmma_f32_16x16x32_f16(false, Qa[c].v, false, B0.v,
                                                        (short)0, S0, false, false);
            S1 = __builtin_amdgcn_wmma_f32_16x16x32_f16(false, Qa[c].v, false, B1.v,
                                                        (short)0, S1, false, false);
        }
        #pragma unroll
        for (int r = 0; r < 8; ++r) {
            float s0 = S0[r], s1 = S1[r];
            if (needMask) {
                const int qi = qBase + r + hiHalf * 8;
                if (j0 + laneN      > qi) s0 = -3.0e38f;
                if (j0 + 16 + laneN > qi) s1 = -3.0e38f;
            }
            const float p0 = __builtin_amdgcn_exp2f(s0);
            const float p1 = __builtin_amdgcn_exp2f(s1);
            const int m = r + hiHalf * 8;
            Psh[wid][m * 32 + laneN]      = (_Float16)p0;
            Psh[wid][m * 32 + 16 + laneN] = (_Float16)p1;
        }
        AFrag Pa;
        Pa.h[0] = *(const v8h*)&Psh[wid][laneN * 32 + 8 * hiHalf];
        Pa.h[1] = *(const v8h*)&Psh[wid][laneN * 32 + 16 + 8 * hiHalf];

        Lacc = __builtin_amdgcn_wmma_f32_16x16x32_f16(false, Pa.v, false, OnesB.v,
                                                      (short)0, Lacc, false, false);
        #pragma unroll
        for (int dt = 0; dt < 4; ++dt) {
            AFrag Vb;
            const int vr = (dt * 16 + laneN) * VSTRH + 8 * hiHalf;
            Vb.h[0] = *(const v8h*)&VshT[vr];
            Vb.h[1] = *(const v8h*)&VshT[vr + 16];
            Oacc[dt] = __builtin_amdgcn_wmma_f32_16x16x32_f16(false, Pa.v, false, Vb.v,
                                                              (short)0, Oacc[dt],
                                                              false, false);
        }
    }

    #pragma unroll
    for (int r = 0; r < 8; ++r) {
        const int qi  = qBase + r + hiHalf * 8;
        const float inv = 1.0f / Lacc[r];
        float y[4];
        float ss = 0.0f;
        #pragma unroll
        for (int dt = 0; dt < 4; ++dt) {
            y[dt] = Oacc[dt][r] * inv;
            ss += y[dt] * y[dt];
        }
        #pragma unroll
        for (int sft = 1; sft < 16; sft <<= 1)
            ss += __shfl_xor(ss, sft, 32);
        const float t = sqrtf(ss + 1.0f);
        float* orow = Oh + (size_t)qi * DL;
        #pragma unroll
        for (int dt = 0; dt < 4; ++dt)
            orow[1 + dt * 16 + laneN] = y[dt];
        if (laneN == 0) orow[0] = t;
    }
}

extern "C" void kernel_launch(void* const* d_in, const int* in_sizes, int n_in,
                              void* d_out, int out_size, void* d_ws, size_t ws_size,
                              hipStream_t stream) {
    (void)in_sizes; (void)n_in; (void)out_size;
    const float* Q  = (const float*)d_in[0];
    const float* K  = (const float*)d_in[1];
    const float* V  = (const float*)d_in[2];
    const int* causal = (const int*)d_in[3];
    float* out = (float*)d_out;

    dim3 grid(NSEQ / 128, BH);
    dim3 block(256);
    if (ws_size >= WS_NEED) {
        _Float16* K16  = (_Float16*)d_ws;
        _Float16* V16T = K16 + K16_ELEMS;
        dim3 pgrid(NSEQ / 32, BH);
        lorentz_prep_f16<<<pgrid, block, 0, stream>>>(K, V, K16, V16T);
        lorentz_flash_attn_tdm<<<grid, block, 0, stream>>>(Q, K16, V16T, causal, out);
    } else {
        lorentz_flash_attn_fallback<<<grid, block, 0, stream>>>(Q, K, V, causal, out);
    }
}
